// Attention_52767968198904
// MI455X (gfx1250) — matrix operand order repaired and checked by reading, not compiled, not run on hardware
//
#include <hip/hip_runtime.h>
#include <hip/hip_bf16.h>

// ---------------------------------------------------------------------------
// Problem constants (match reference)
// ---------------------------------------------------------------------------
#define B_    32
#define S_    256
#define T_    (B_ * S_)          // 8192 tokens
#define DIM_  2048
#define NH_   16
#define HD_   128
#define NQKV_ 6144               // (NH + 2*NKV) * HD

typedef __bf16 bf16;
typedef __attribute__((ext_vector_type(16))) __bf16 v16bf;
typedef __attribute__((ext_vector_type(8)))  float  v8f;

union V16U { v16bf v; uint4 u[2]; };   // 32B fragment (16 bf16)
union U8   { uint4 u;  bf16 e[8]; };   // 16B (8 bf16)

__device__ __forceinline__ bf16 to_bf16(float f) {
    unsigned u = __builtin_bit_cast(unsigned, f);
    u += 0x7FFFu + ((u >> 16) & 1u);               // round-to-nearest-even
    unsigned short h = (unsigned short)(u >> 16);
    return __builtin_bit_cast(bf16, h);
}
__device__ __forceinline__ float bf16_to_f32(bf16 h) {
    unsigned short s = __builtin_bit_cast(unsigned short, h);
    unsigned u = ((unsigned)s) << 16;
    return __builtin_bit_cast(float, u);
}

__device__ __forceinline__ v8f wmma_bf16(const V16U& a, const V16U& b, v8f c) {
    return __builtin_amdgcn_wmma_f32_16x16x32_bf16(
        /*neg_a=*/false, a.v, /*neg_b=*/false, b.v,
        /*c_mod=*/(short)0, c, /*reuse_a=*/false, /*reuse_b=*/false);
}

__device__ __forceinline__ void storeOut(float* p, float v) { *p = v; }
__device__ __forceinline__ void storeOut(bf16*  p, float v) { *p = to_bf16(v); }

// ---------------------------------------------------------------------------
// fp32 -> bf16 conversion (grid-stride)
// ---------------------------------------------------------------------------
__global__ void cvt_bf16(const float* __restrict__ src, bf16* __restrict__ dst,
                         size_t n) {
    size_t i = (size_t)blockIdx.x * blockDim.x + threadIdx.x;
    size_t stride = (size_t)gridDim.x * blockDim.x;
    for (; i < n; i += stride) dst[i] = to_bf16(src[i]);
}

// ---------------------------------------------------------------------------
// C[M,N] = A[M,K] * B[N,K]^T   (bf16 inputs, fp32 accumulate, OutT output)
// Block tile 128x128, 8 waves (4x2), each wave 32x64 (2x4 WMMA tiles).
// Double-buffered LDS: global loads for step k+1 are issued before the WMMAs
// of step k, so HBM/L2 latency overlaps the matrix pipe (one barrier/step).
// ---------------------------------------------------------------------------
template <typename OutT>
__global__ __launch_bounds__(256) void gemm_abt(
    const bf16* __restrict__ A, const bf16* __restrict__ Bm,
    OutT* __restrict__ C, int M, int N, int K)
{
    __shared__ bf16 As[2][128][40];   // +8 pad: 80B row pitch, 16B aligned
    __shared__ bf16 Bs[2][128][40];

    const int tid  = threadIdx.x;
    const int lane = tid & 31, wid = tid >> 5;
    const int half = lane >> 4, l16 = lane & 15;
    const int m0 = blockIdx.y * 128, n0 = blockIdx.x * 128;
    const int mrow0 = (wid >> 1) * 32;   // wave row  (4 rows of waves)
    const int ncol0 = (wid & 1) * 64;    // wave col  (2 cols of waves)
    const int ldr = tid >> 1;            // 0..127  tile row for coop load
    const int ldk = (tid & 1) * 16;      // 0 or 16 k-offset (16 bf16 = 32B)

    v8f acc[2][4];
    #pragma unroll
    for (int mi = 0; mi < 2; ++mi)
        #pragma unroll
        for (int ni = 0; ni < 4; ++ni)
            #pragma unroll
            for (int r = 0; r < 8; ++r) acc[mi][ni][r] = 0.0f;

    const int nsteps = K >> 5;
    uint4 a0, a1, b0, b1;

    // prologue: fill buffer 0
    {
        const uint4* ga = (const uint4*)(A  + (size_t)(m0 + ldr) * K + ldk);
        a0 = ga[0]; a1 = ga[1];
        const uint4* gb = (const uint4*)(Bm + (size_t)(n0 + ldr) * K + ldk);
        b0 = gb[0]; b1 = gb[1];
        *(uint4*)&As[0][ldr][ldk]     = a0;  *(uint4*)&As[0][ldr][ldk + 8] = a1;
        *(uint4*)&Bs[0][ldr][ldk]     = b0;  *(uint4*)&Bs[0][ldr][ldk + 8] = b1;
    }
    __syncthreads();

    for (int step = 0; step < nsteps; ++step) {
        const int cur = step & 1, nxt = cur ^ 1;

        // issue global loads for the NEXT tile before computing this one
        if (step + 1 < nsteps) {
            const int k0 = (step + 1) << 5;
            const uint4* ga = (const uint4*)(A  + (size_t)(m0 + ldr) * K + k0 + ldk);
            a0 = ga[0]; a1 = ga[1];
            const uint4* gb = (const uint4*)(Bm + (size_t)(n0 + ldr) * K + k0 + ldk);
            b0 = gb[0]; b1 = gb[1];
        }

        // fragments + compute from current buffer
        V16U a[2], b[4];
        #pragma unroll
        for (int mi = 0; mi < 2; ++mi) {
            int m = mrow0 + mi * 16 + l16;
            a[mi].u[0] = *(const uint4*)&As[cur][m][8 * half];       // k = 8h..8h+7
            a[mi].u[1] = *(const uint4*)&As[cur][m][16 + 8 * half];  // k = 16+8h..
        }
        #pragma unroll
        for (int ni = 0; ni < 4; ++ni) {
            int n = ncol0 + ni * 16 + l16;
            b[ni].u[0] = *(const uint4*)&Bs[cur][n][8 * half];       // k = 8h..8h+7
            b[ni].u[1] = *(const uint4*)&Bs[cur][n][16 + 8 * half];
        }
        #pragma unroll
        for (int mi = 0; mi < 2; ++mi)
            #pragma unroll
            for (int ni = 0; ni < 4; ++ni)
                acc[mi][ni] = wmma_bf16(a[mi], b[ni], acc[mi][ni]);

        // park the prefetched tile in the alternate buffer
        if (step + 1 < nsteps) {
            *(uint4*)&As[nxt][ldr][ldk]     = a0;  *(uint4*)&As[nxt][ldr][ldk + 8] = a1;
            *(uint4*)&Bs[nxt][ldr][ldk]     = b0;  *(uint4*)&Bs[nxt][ldr][ldk + 8] = b1;
            __syncthreads();
        }
    }

    // D layout: n = lane&15, m = r + 8*(lane>>4)
    #pragma unroll
    for (int mi = 0; mi < 2; ++mi)
        #pragma unroll
        for (int ni = 0; ni < 4; ++ni)
            #pragma unroll
            for (int r = 0; r < 8; ++r) {
                int m = m0 + mrow0 + mi * 16 + r + 8 * half;
                int n = n0 + ncol0 + ni * 16 + l16;
                storeOut(C + (size_t)m * N + n, acc[mi][ni][r]);
            }
}

// ---------------------------------------------------------------------------
// RoPE (2D, computed on the fly from positions) + scatter to [b][h][s][d]
// ---------------------------------------------------------------------------
__global__ __launch_bounds__(256) void rope_scatter(
    const bf16* __restrict__ qkv, const int* __restrict__ pos,
    bf16* __restrict__ Qb, bf16* __restrict__ Kb, bf16* __restrict__ Vb)
{
    size_t g = (size_t)blockIdx.x * blockDim.x + threadIdx.x;
    if (g >= (size_t)T_ * (NQKV_ / 2)) return;
    int n2 = (int)(g % (NQKV_ / 2));
    int t  = (int)(g / (NQKV_ / 2));
    int n  = n2 * 2;
    int b = t / S_, s = t % S_;
    float x0 = bf16_to_f32(qkv[(size_t)t * NQKV_ + n]);
    float x1 = bf16_to_f32(qkv[(size_t)t * NQKV_ + n + 1]);

    if (n < 2 * DIM_) {                       // q or k -> rope
        int isK = n >= DIM_;
        int nn = isK ? n - DIM_ : n;
        int h = nn >> 7, d = nn & 127, j = d >> 1;
        int p = pos[t];
        float c = 0.0f, sn = 0.0f;
        if (p != 0) {                          // p==0 (cls) -> zeros
            int gidx = p - 1;
            int coord = (j < 32) ? (gidx >> 4) : (gidx & 15);
            int jj = (j < 32) ? j : j - 32;
            float f = (float)coord * __powf(10000.0f, -(float)jj * (1.0f / 32.0f));
            c = __cosf(f);  sn = __sinf(f);
        }
        float o0 = x0 * c - x1 * sn;
        float o1 = x1 * c + x0 * sn;
        bf16* dst = (isK ? Kb : Qb) + ((size_t)(b * NH_ + h) * S_ + s) * HD_ + d;
        dst[0] = to_bf16(o0);  dst[1] = to_bf16(o1);
    } else {                                   // v -> copy
        int nn = n - 2 * DIM_;
        int h = nn >> 7, d = nn & 127;
        bf16* dst = Vb + ((size_t)(b * NH_ + h) * S_ + s) * HD_ + d;
        dst[0] = qkv[(size_t)t * NQKV_ + n];
        dst[1] = qkv[(size_t)t * NQKV_ + n + 1];
    }
}

// ---------------------------------------------------------------------------
// Causal attention, one workgroup per (b,h). K and V^T staged in dynamic LDS.
// ---------------------------------------------------------------------------
#define KPITCH 136   // 128 + 8 pad (bf16 elems)
#define VPITCH 264   // 256 + 8 pad (bf16 elems)

// One 32-key online-softmax step. MASKED is true only for the diagonal step;
// interior steps skip all causal compare/select VALU work.
template <bool MASKED>
__device__ __forceinline__ void attn_step(
    int k0, int q0, int half, int l16,
    const V16U (&aq)[4], const bf16* __restrict__ Ks,
    const bf16* __restrict__ Vt, bf16* __restrict__ Pw,
    float (&mrun)[8], float (&lrun)[8], v8f (&acc)[8])
{
    const float scale = 0.08838834764831845f;   // 1/sqrt(128)
    v8f s0, s1;
    #pragma unroll
    for (int r = 0; r < 8; ++r) { s0[r] = 0.0f; s1[r] = 0.0f; }

    #pragma unroll
    for (int c = 0; c < 4; ++c) {
        V16U bk0, bk1;    // B layout: n = key = lane&15, k = d-chunk
        const bf16* kr0 = Ks + (k0 + l16) * KPITCH + c * 32 + 8 * half;
        bk0.u[0] = *(const uint4*)(kr0);
        bk0.u[1] = *(const uint4*)(kr0 + 16);
        const bf16* kr1 = Ks + (k0 + 16 + l16) * KPITCH + c * 32 + 8 * half;
        bk1.u[0] = *(const uint4*)(kr1);
        bk1.u[1] = *(const uint4*)(kr1 + 16);
        s0 = wmma_bf16(aq[c], bk0, s0);
        s1 = wmma_bf16(aq[c], bk1, s1);
    }

    #pragma unroll
    for (int r = 0; r < 8; ++r) {
        const int m = q0 + r + 8 * half;
        float x0 = s0[r] * scale;
        float x1 = s1[r] * scale;
        if (MASKED) {
            if (k0 + l16 > m)      x0 = -3.0e38f;
            if (k0 + 16 + l16 > m) x1 = -3.0e38f;
        }
        float mx = fmaxf(x0, x1);
        #pragma unroll
        for (int off = 8; off >= 1; off >>= 1)
            mx = fmaxf(mx, __shfl_xor(mx, off, 16));
        float newm = fmaxf(mrun[r], mx);
        float corr = __expf(mrun[r] - newm);
        float p0 = __expf(x0 - newm), p1 = __expf(x1 - newm);
        float rs = p0 + p1;
        #pragma unroll
        for (int off = 8; off >= 1; off >>= 1)
            rs += __shfl_xor(rs, off, 16);
        lrun[r] = lrun[r] * corr + rs;
        mrun[r] = newm;
        #pragma unroll
        for (int nt = 0; nt < 8; ++nt) acc[nt][r] *= corr;
        // stash P (D layout -> LDS [m][k] for A-layout reload)
        Pw[(r + 8 * half) * 32 + l16]      = to_bf16(p0);
        Pw[(r + 8 * half) * 32 + 16 + l16] = to_bf16(p1);
    }
    asm volatile("s_wait_dscnt 0" ::: "memory");

    V16U pa;   // A layout: m = lane&15, k per half-wave chunks
    pa.u[0] = *(const uint4*)(Pw + l16 * 32 + 8 * half);
    pa.u[1] = *(const uint4*)(Pw + l16 * 32 + 16 + 8 * half);

    #pragma unroll
    for (int nt = 0; nt < 8; ++nt) {
        V16U bv;  // B layout: n = d = lane&15, k = key (contig in Vt)
        const bf16* vr = Vt + (nt * 16 + l16) * VPITCH + k0 + 8 * half;
        bv.u[0] = *(const uint4*)(vr);
        bv.u[1] = *(const uint4*)(vr + 16);
        acc[nt] = wmma_bf16(pa, bv, acc[nt]);
    }
}

__global__ __launch_bounds__(256) void attn_kernel(
    const bf16* __restrict__ Q, const bf16* __restrict__ K,
    const bf16* __restrict__ V, bf16* __restrict__ O)
{
    const int bh = blockIdx.x;
    const int b = bh / NH_, h = bh % NH_;
    const int tid = threadIdx.x, lane = tid & 31, wid = tid >> 5;
    const int half = lane >> 4, l16 = lane & 15;

    extern __shared__ __align__(16) char smem_raw[];
    bf16* Ks = (bf16*)smem_raw;                 // [256][KPITCH] row-major (s,d)
    bf16* Vt = Ks + 256 * KPITCH;               // [128][VPITCH] transposed (d,s)
    bf16* Pw = Vt + 128 * VPITCH + wid * (16 * 32); // per-wave P scratch

    // cooperative load: one K row + one V row (transposed) per thread
    {
        int s = tid;
        const uint4* ksrc = (const uint4*)(K + ((size_t)(b * NH_ + h) * S_ + s) * HD_);
        uint4* kdst = (uint4*)(Ks + s * KPITCH);
        #pragma unroll
        for (int i = 0; i < 16; ++i) kdst[i] = ksrc[i];
        const uint4* vsrc = (const uint4*)(V + ((size_t)(b * NH_ + h) * S_ + s) * HD_);
        #pragma unroll
        for (int i = 0; i < 16; ++i) {
            U8 t8; t8.u = vsrc[i];
            #pragma unroll
            for (int j = 0; j < 8; ++j) Vt[(i * 8 + j) * VPITCH + s] = t8.e[j];
        }
    }
    __syncthreads();

    for (int phase = 0; phase < 2; ++phase) {
        const int qt = wid * 2 + phase;
        const int q0 = qt * 16;

        // Q fragments: 4 chunks of k=32 over d=128 (A layout: m = lane&15)
        V16U aq[4];
        const bf16* qrow = Q + ((size_t)(b * NH_ + h) * S_ + q0 + l16) * HD_;
        #pragma unroll
        for (int c = 0; c < 4; ++c) {
            aq[c].u[0] = *(const uint4*)(qrow + c * 32 + 8 * half);
            aq[c].u[1] = *(const uint4*)(qrow + c * 32 + 16 + 8 * half);
        }

        float mrun[8], lrun[8];
        v8f acc[8];
        #pragma unroll
        for (int r = 0; r < 8; ++r) { mrun[r] = -3.0e38f; lrun[r] = 0.0f; }
        #pragma unroll
        for (int nt = 0; nt < 8; ++nt)
            #pragma unroll
            for (int r = 0; r < 8; ++r) acc[nt][r] = 0.0f;

        const int tpmax = qt >> 1;              // causal: keys [0, q0+15]
        for (int tp = 0; tp < tpmax; ++tp)      // interior steps: no masking
            attn_step<false>(tp * 32, q0, half, l16, aq, Ks, Vt, Pw,
                             mrun, lrun, acc);
        attn_step<true>(tpmax * 32, q0, half, l16, aq, Ks, Vt, Pw,
                        mrun, lrun, acc);       // diagonal step: causal mask

        // normalize, write O at [b][s][h][d] (= row-major (T, 2048))
        #pragma unroll
        for (int r = 0; r < 8; ++r) {
            const int m = q0 + r + 8 * half;
            const float inv = 1.0f / lrun[r];
            bf16* orow = O + ((size_t)(b * S_ + m) * NH_ + h) * HD_;
            #pragma unroll
            for (int nt = 0; nt < 8; ++nt)
                orow[nt * 16 + l16] = to_bf16(acc[nt][r] * inv);
        }
    }
}

// ---------------------------------------------------------------------------
// Host launcher
// ---------------------------------------------------------------------------
extern "C" void kernel_launch(void* const* d_in, const int* in_sizes, int n_in,
                              void* d_out, int out_size, void* d_ws, size_t ws_size,
                              hipStream_t stream) {
    const float* x         = (const float*)d_in[0];
    const int*   positions = (const int*)  d_in[1];
    const float* wqkv      = (const float*)d_in[2];
    const float* wo        = (const float*)d_in[3];
    float*       out       = (float*)d_out;
    char*        ws        = (char*)d_ws;

    // workspace layout (bytes)
    bf16* xb   = (bf16*)(ws);                 //  32 MB  x bf16
    bf16* wqb  = (bf16*)(ws + 33554432);      //  24 MB  wqkv bf16
    bf16* wob  = (bf16*)(ws + 58720256);      //   8 MB  wo bf16
    bf16* qkvb = (bf16*)(ws + 67108864);      //  96 MB  qkv bf16 (T x 6144)
    bf16* Qb   = (bf16*)(ws + 167772160);     //  32 MB  [b][h][s][d]
    bf16* Kb   = (bf16*)(ws + 201326592);     //  32 MB
    bf16* Vb   = (bf16*)(ws + 234881024);     //  32 MB
    bf16* Ob   = (bf16*)(ws + 268435456);     //  32 MB  [b][s][h][d]

    cvt_bf16<<<4096, 256, 0, stream>>>(x,    xb,  (size_t)T_ * DIM_);
    cvt_bf16<<<4096, 256, 0, stream>>>(wqkv, wqb, (size_t)NQKV_ * DIM_);
    cvt_bf16<<<2048, 256, 0, stream>>>(wo,   wob, (size_t)DIM_ * DIM_);

    // qkv = x @ wqkv^T  (8192 x 6144 x 2048)
    gemm_abt<bf16><<<dim3(NQKV_ / 128, T_ / 128), 256, 0, stream>>>(
        xb, wqb, qkvb, T_, NQKV_, DIM_);

    rope_scatter<<<(T_ * (NQKV_ / 2)) / 256, 256, 0, stream>>>(
        qkvb, positions, Qb, Kb, Vb);

    size_t smem = (size_t)(256 * KPITCH + 128 * VPITCH + 8 * 16 * 32) * sizeof(bf16);
    attn_kernel<<<B_ * NH_, 256, smem, stream>>>(Qb, Kb, Vb, Ob);

    // out = attn @ wo^T  (8192 x 2048 x 2048), fp32 output
    gemm_abt<float><<<dim3(DIM_ / 128, T_ / 128), 256, 0, stream>>>(
        Ob, wob, out, T_, DIM_, DIM_);
}
